// UNet5_38981123179037
// MI455X (gfx1250) — matrix operand order repaired and checked by reading, not compiled, not run on hardware
//
#include <hip/hip_runtime.h>

typedef __attribute__((ext_vector_type(16))) _Float16 v16h;
typedef __attribute__((ext_vector_type(8)))  float    v8f;
typedef __attribute__((ext_vector_type(4)))  float    v4f;

#define EPSV 1e-8f

// ---- LDS layout (bank-conflict padded, all 16B aligned) ----
#define BV_STRIDE_H 2056            // halves per voxel row block in sB16 (4112 B)
#define BV_STRIDE_B 4112
#define OFF_P       65792           // sB16 = 16*4112 = 65792 B
#define P_WSTRIDE   5120            // 64 rows * 80 B
#define P_ROW       80
#define OFF_F       86272           // OFF_P + 4*5120
#define F_WSTRIDE   4608            // 32 rows * 144 B
#define F_ROW       144
#define OFF_DEN     104704          // OFF_F + 4*4608
#define OFF_PART    104768
#define SMEM_BYTES  108864          // OFF_PART + 4*16*16*4

__device__ __forceinline__ v16h ld_frag(const char* p0, const char* p1) {
  union { v4f f[2]; v16h h; } u;
  u.f[0] = *(const v4f*)p0;
  u.f[1] = *(const v4f*)p1;
  return u.h;
}

// f32 row (16 consecutive floats) -> f16 B-operand fragment
__device__ __forceinline__ v16h cvt_frag(const float* frow) {
  v4f f0 = *(const v4f*)(frow);
  v4f f1 = *(const v4f*)(frow + 4);
  v4f f2 = *(const v4f*)(frow + 8);
  v4f f3 = *(const v4f*)(frow + 12);
  v16h h;
  #pragma unroll
  for (int i = 0; i < 4; ++i) {
    h[i]      = (_Float16)f0[i];
    h[4 + i]  = (_Float16)f1[i];
    h[8 + i]  = (_Float16)f2[i];
    h[12 + i] = (_Float16)f3[i];
  }
  return h;
}

__global__ void __launch_bounds__(256) wconv_f16(const float* __restrict__ W,
                                                 _Float16* __restrict__ Wh, int n) {
  int i = blockIdx.x * 256 + threadIdx.x;
  if (i < n) Wh[i] = (_Float16)W[i];
}

// One workgroup (4 wave32) handles 16 output voxels.
// Phase 1 (per wave, 4 voxels): B_v[64,32] = P[64,edges] @ F[edges,32] via WMMA.
//   P = trilinear coefs (f16 LDS, conflict-free column-per-lane scatter).
//   F = neighbor feature rows DMA'd with GLOBAL_LOAD_ASYNC_TO_LDS_B128 (ASYNCcnt).
// Phase 2: Y[16,32] = Btile[16,2048] @ Wf16[2048,32] via WMMA, K and N each split in 2 over 4 waves.
// Phase 3: cross-wave reduce + normalize + bias + relu.
__global__ void __launch_bounds__(128) cconv_fused(
    const float* __restrict__ feats,
    const float* __restrict__ inp_points,
    const float* __restrict__ out_points,
    const float* __restrict__ out_extents,
    const float* __restrict__ scale_compat,
    const long long* __restrict__ nbr_index,
    const long long* __restrict__ row_splits,
    const float* __restrict__ nbr_dist,
    const _Float16* __restrict__ Wh,
    const float* __restrict__ bias,
    float* __restrict__ out) {
  extern __shared__ char smem[];
  const int lane = threadIdx.x & 31;
  const int w    = threadIdx.x >> 5;
  const int half = (lane < 16) ? 0 : 1;
  const int m16  = lane & 15;

  _Float16* sB16 = (_Float16*)smem;                      // [16][64][32] padded
  char*     sPb  = smem + OFF_P + w * P_WSTRIDE;         // per-wave P [64 rows x 80B]
  char*     sFb  = smem + OFF_F + w * F_WSTRIDE;         // per-wave F [32 rows x 144B] f32
  float*    sDen = (float*)(smem + OFF_DEN);             // [16]
  float*    sPart= (float*)(smem + OFF_PART);            // [4][16][16] f32

  const float inv_ext2 = 2.0f / out_extents[0];
  const int tile = blockIdx.x;
  const v8f z8 = {0.f,0.f,0.f,0.f,0.f,0.f,0.f,0.f};
  const v4f z4 = {0.f,0.f,0.f,0.f};

  // Warm W (f16, 128KB) into cache for phase 2.
  #pragma unroll
  for (int i = 0; i < 8; ++i)
    __builtin_prefetch((const char*)Wh + ((unsigned)(i * 128 + threadIdx.x) * 128u), 0, 0);

  // ---------------- Phase 1: build B tile ----------------
  for (int vl = 0; vl < 4; ++vl) {
    const int vloc = w * 4 + vl;
    const int vg = tile * 16 + vloc;
    const long long r0 = row_splits[vg];
    int cnt = (int)(row_splits[vg + 1] - r0);
    cnt = __builtin_amdgcn_readfirstlane(cnt);           // wave-uniform loop bound
    const float ox = out_points[vg*3+0];
    const float oy = out_points[vg*3+1];
    const float oz = out_points[vg*3+2];

    v8f acc[4][2];
    #pragma unroll
    for (int i = 0; i < 4; ++i) { acc[i][0] = z8; acc[i][1] = z8; }
    float dlane = 0.0f;

    for (int e0 = 0; e0 < cnt; e0 += 32) {
      // zero P (5120 B = 10 x b128 per lane)
      #pragma unroll
      for (int z = 0; z < 10; ++z) ((v4f*)sPb)[z * 32 + lane] = z4;
      asm volatile("" ::: "memory");

      // ---- lane = edge: geometry, importance, trilinear coefs ----
      const int e = e0 + lane;
      const bool act = e < cnt;
      const int ec = act ? e : (cnt - 1);
      const size_t eg = (size_t)(r0 + (long long)ec);
      const int idx = (int)nbr_index[eg];
      const float dist = nbr_dist[eg];
      const float sc = scale_compat[eg];
      float win = 1.0f - dist * dist;
      win = fminf(fmaxf(win, 0.0f), 1.0f);
      float a = sc * win * win * win;
      a = act ? a : 0.0f;
      dlane += a;

      // async DMA this lane's neighbor feature row (128B) into LDS row `lane`
      {
        const unsigned ldsF = (unsigned)(OFF_F + w * F_WSTRIDE + lane * F_ROW);
        const unsigned goff = (unsigned)idx * 128u;      // feats row byte offset
        asm volatile(
          "global_load_async_to_lds_b128 %0, %1, %2 offset:0\n\t"
          "global_load_async_to_lds_b128 %0, %1, %2 offset:16\n\t"
          "global_load_async_to_lds_b128 %0, %1, %2 offset:32\n\t"
          "global_load_async_to_lds_b128 %0, %1, %2 offset:48\n\t"
          "global_load_async_to_lds_b128 %0, %1, %2 offset:64\n\t"
          "global_load_async_to_lds_b128 %0, %1, %2 offset:80\n\t"
          "global_load_async_to_lds_b128 %0, %1, %2 offset:96\n\t"
          "global_load_async_to_lds_b128 %0, %1, %2 offset:112\n\t"
          :: "v"(ldsF), "v"(goff), "s"(feats) : "memory");
      }

      const float rx = (inp_points[idx*3+0] - ox) * inv_ext2;
      const float ry = (inp_points[idx*3+1] - oy) * inv_ext2;
      const float rz = (inp_points[idx*3+2] - oz) * inv_ext2;
      const float n2 = sqrtf(rx*rx + ry*ry + rz*rz + 1e-12f);
      const float ninf = fmaxf(fabsf(rx), fmaxf(fabsf(ry), fabsf(rz)));
      const float bsc = (ninf > 1e-8f) ? (n2 / fmaxf(ninf, 1e-8f)) : 0.0f;
      const float cx = fminf(1.f, fmaxf(-1.f, rx * bsc));
      const float cy = fminf(1.f, fmaxf(-1.f, ry * bsc));
      const float cz = fminf(1.f, fmaxf(-1.f, rz * bsc));
      const float uxv = fminf(3.f, fmaxf(0.f, (cx + 1.f) * 1.5f));
      const float uyv = fminf(3.f, fmaxf(0.f, (cy + 1.f) * 1.5f));
      const float uzv = fminf(3.f, fmaxf(0.f, (cz + 1.f) * 1.5f));
      const float fx = fminf(floorf(uxv), 2.f);
      const float fy = fminf(floorf(uyv), 2.f);
      const float fz = fminf(floorf(uzv), 2.f);
      const int ix = (int)fx, iy = (int)fy, iz = (int)fz;
      const float tx = uxv - fx, ty = uyv - fy, tz = uzv - fz;

      // scatter 8 corner coefficients into P (column-per-lane: conflict-free)
      #pragma unroll
      for (int dx = 0; dx < 2; ++dx) {
        const float wx = dx ? tx : (1.f - tx);
        #pragma unroll
        for (int dy = 0; dy < 2; ++dy) {
          const float wy = dy ? ty : (1.f - ty);
          #pragma unroll
          for (int dz = 0; dz < 2; ++dz) {
            const float wz = dz ? tz : (1.f - tz);
            const int kf = ((ix + dx) * 4 + (iy + dy)) * 4 + (iz + dz);
            *(_Float16*)(sPb + kf * P_ROW + lane * 2) = (_Float16)(a * wx * wy * wz);
          }
        }
      }

      // wait for the feature DMA, then build the two F fragments (column = this lane, k = edge rows)
      asm volatile("s_wait_asynccnt 0x0" ::: "memory");
      v16h bf0, bf1;
      #pragma unroll
      for (int s = 0; s < 16; ++s) {
        const int kk = (s < 8) ? (8 * half + s) : (16 + 8 * half + (s - 8));
        const float* fr = (const float*)(sFb + kk * F_ROW);
        bf0[s] = (_Float16)fr[m16];
        bf1[s] = (_Float16)fr[16 + m16];
      }

      // ---- B_v += P @ F  (4 M-tiles x 2 N-tiles of 16x16x32) ----
      #pragma unroll
      for (int mt = 0; mt < 4; ++mt) {
        const char* ab = sPb + (mt * 16 + m16) * P_ROW + half * 16;
        const v16h af = ld_frag(ab, ab + 32);
        acc[mt][0] = __builtin_amdgcn_wmma_f32_16x16x32_f16(
            false, af, false, bf0, (short)0, acc[mt][0], false, false);
        acc[mt][1] = __builtin_amdgcn_wmma_f32_16x16x32_f16(
            false, af, false, bf1, (short)0, acc[mt][1], false, false);
      }
    } // edge chunks

    // denominator (segment sum of a), once per voxel
    float asum = dlane;
    #pragma unroll
    for (int s = 16; s > 0; s >>= 1) asum += __shfl_xor(asum, s, 32);
    if (lane == 0) sDen[vloc] = asum;

    // D (f32) -> f16 B tile in LDS. D layout: M = j + 8*half, N = m16
    #pragma unroll
    for (int mt = 0; mt < 4; ++mt) {
      #pragma unroll
      for (int nt = 0; nt < 2; ++nt) {
        #pragma unroll
        for (int j = 0; j < 8; ++j) {
          const int bin = mt * 16 + j + half * 8;
          const int c = nt * 16 + m16;
          sB16[vloc * BV_STRIDE_H + bin * 32 + c] = (_Float16)acc[mt][nt][j];
        }
      }
    }
  } // voxels
  __syncthreads();

  // ---------------- Phase 2: Y = Btile @ W (K split across waves) ----------------
  {
    const int nt = w & 1;        // cout half
    const int kh = w >> 1;       // bin half
    v8f y = z8;
    for (int kb = 0; kb < 32; ++kb) {
      const int bin = kh * 32 + kb;
      const char* ab = (const char*)sB16 + m16 * BV_STRIDE_B + bin * 64 + half * 16;
      const v16h af = ld_frag(ab, ab + 32);
      const _Float16* wb = Wh + ((size_t)bin * 1024 + nt * 16 + m16);
      v16h bf;
      #pragma unroll
      for (int s = 0; s < 16; ++s) {
        const int kk = (s < 8) ? (8 * half + s) : (16 + 8 * half + (s - 8));
        bf[s] = wb[kk * 32];
      }
      y = __builtin_amdgcn_wmma_f32_16x16x32_f16(false, af, false, bf, (short)0, y,
                                                 false, false);
    }
    #pragma unroll
    for (int j = 0; j < 8; ++j) {
      const int m = j + half * 8;
      sPart[(w * 16 + m) * 16 + m16] = y[j];
    }
  }
  __syncthreads();

  // ---------------- Phase 3: reduce partials, normalize, bias, relu ----------------
  #pragma unroll
  for (int q = 0; q < 4; ++q) {
    const int oidx = q * 128 + threadIdx.x;   // 0..511 = 16 voxels x 32 cout
    const int m = oidx >> 5;
    const int c = oidx & 31;
    const int nt2 = c >> 4;
    const int n = c & 15;
    float y = sPart[(nt2 * 16 + m) * 16 + n] + sPart[((nt2 + 2) * 16 + m) * 16 + n];
    y = y / fmaxf(sDen[m], EPSV) + bias[c];
    out[((size_t)tile * 16 + m) * 32 + c] = fmaxf(y, 0.0f);
  }
}

extern "C" void kernel_launch(void* const* d_in, const int* in_sizes, int n_in,
                              void* d_out, int out_size, void* d_ws, size_t ws_size,
                              hipStream_t stream) {
  const float*     feats   = (const float*)d_in[0];
  const float*     inp_pts = (const float*)d_in[1];
  const float*     out_pts = (const float*)d_in[2];
  const float*     extents = (const float*)d_in[3];
  const float*     scomp   = (const float*)d_in[4];
  const long long* nidx    = (const long long*)d_in[5];
  const long long* rsplit  = (const long long*)d_in[6];
  const float*     ndist   = (const float*)d_in[7];
  const float*     W       = (const float*)d_in[8];
  const float*     bias    = (const float*)d_in[9];
  float*           out     = (float*)d_out;
  _Float16*        Wh      = (_Float16*)d_ws;   // 64*32*32 f16 = 128KB

  const int wn = 64 * 32 * 32;
  wconv_f16<<<(wn + 255) / 256, 256, 0, stream>>>(W, Wh, wn);

  const int n_tiles = 65536 / 16;   // 4096 workgroups of 16 voxels
  cconv_fused<<<n_tiles, 128, SMEM_BYTES, stream>>>(
      feats, inp_pts, out_pts, extents, scomp, nidx, rsplit, ndist, Wh, bias, out);
}
